// Decoder_30124900614570
// MI455X (gfx1250) — matrix operand order repaired and checked by reading, not compiled, not run on hardware
//
#include <hip/hip_runtime.h>
#include <hip/hip_bf16.h>
#include <cstdint>

typedef __attribute__((ext_vector_type(16))) _Float16 v16h;
typedef __attribute__((ext_vector_type(8)))  _Float16 v8h;
typedef __attribute__((ext_vector_type(8)))  float    v8f;

#define B_  128
#define S_  2048
#define H_  512
#define E_  256
#define G_  2048      // 4*H
#define KL_ 1024      // 2*H  (LSTM fused K)
#define KF_ 512       // FC K
#define NA_ 1000
#define SPLIT_ 4

// ---------------------------------------------------------------------------
// 1) Repack weights fp32 -> f16:  Wc[l] = [W_ih[l] | W_hh[l]]  (2048 x 1024),
//    fcW16 = [action_fc_W ; target_fc_W]  (2000 x 512)
// ---------------------------------------------------------------------------
__global__ void convert_weights_kernel(const float* __restrict__ W_ih,
                                       const float* __restrict__ W_hh,
                                       const float* __restrict__ aW,
                                       const float* __restrict__ tW,
                                       _Float16* __restrict__ Wc,
                                       _Float16* __restrict__ fcW16) {
    const int total1 = 2 * G_ * KL_;           // 4,194,304
    const int total2 = 2 * NA_ * KF_;          // 1,024,000
    int idx = blockIdx.x * blockDim.x + threadIdx.x;
    if (idx >= total1 + total2) return;
    if (idx < total1) {
        int l = idx / (G_ * KL_);
        int r = idx % (G_ * KL_);
        int g = r / KL_;
        int k = r % KL_;
        float v = (k < H_) ? W_ih[(l * G_ + g) * H_ + k]
                           : W_hh[(l * G_ + g) * H_ + (k - H_)];
        Wc[idx] = (_Float16)v;
    } else {
        int j = idx - total1;
        float v = (j < NA_ * KF_) ? aW[j] : tW[j - NA_ * KF_];
        fcW16[j] = (_Float16)v;
    }
}

// ---------------------------------------------------------------------------
// 2) A0 = [action_emb[x0] | target_emb[x1] | hidden[0]] (128 x 1024, f16)
//    A1 upper half = hidden[1]
// ---------------------------------------------------------------------------
__global__ void build_A_kernel(const int* __restrict__ x,
                               const float* __restrict__ aemb,
                               const float* __restrict__ temb,
                               const float* __restrict__ hidden,
                               _Float16* __restrict__ A0,
                               _Float16* __restrict__ A1) {
    const int totalA0 = B_ * KL_;              // 131072
    const int totalA1 = B_ * H_;               // 65536
    int idx = blockIdx.x * blockDim.x + threadIdx.x;
    if (idx >= totalA0 + totalA1) return;
    if (idx < totalA0) {
        int b = idx >> 10, k = idx & 1023;
        float v;
        if (k < E_)           v = aemb[x[b] * E_ + k];
        else if (k < 2 * E_)  v = temb[x[B_ + b] * E_ + (k - E_)];
        else                  v = hidden[b * H_ + (k - 2 * E_)];      // layer 0
        A0[idx] = (_Float16)v;
    } else {
        int j = idx - totalA0;
        int b = j >> 9, k = j & 511;
        A1[b * KL_ + H_ + k] = (_Float16)hidden[B_ * H_ + b * H_ + k]; // layer 1
    }
}

// ---------------------------------------------------------------------------
// WMMA helper macro: concatenate two v8h into v16h
// ---------------------------------------------------------------------------
#define CAT16(lo, hi) __builtin_shufflevector(lo, hi, 0,1,2,3,4,5,6,7,8,9,10,11,12,13,14,15)

// ---------------------------------------------------------------------------
// 3) LSTM GEMM: gates(128,2048) = A(128,1024) @ Wc_l(2048,1024)^T via WMMA f16
//    grid.x = 128 N-tiles; 8 waves per block cover all 8 M-tiles.
// ---------------------------------------------------------------------------
__global__ void lstm_gemm_kernel(const _Float16* __restrict__ A,
                                 const _Float16* __restrict__ Wc_l,
                                 float* __restrict__ gates) {
    int wave = threadIdx.x >> 5;
    int lane = threadIdx.x & 31;
    int ntile = blockIdx.x;
    int mtile = wave;

    int arow  = mtile * 16 + (lane & 15);
    int akoff = (lane < 16) ? 0 : 8;          // A K-half split (ISA 16-bit A layout)
    int grow  = ntile * 16 + (lane & 15);     // B column n == Wc row g
    int bkoff = (lane < 16) ? 0 : 8;          // B K-half split

    const _Float16* Ap = A    + (size_t)arow * KL_;
    const _Float16* Bp = Wc_l + (size_t)grow * KL_;

    v8f acc = {};
#pragma unroll 4
    for (int kk = 0; kk < KL_; kk += 32) {
        v8h a0 = *(const v8h*)(Ap + kk + akoff);
        v8h a1 = *(const v8h*)(Ap + kk + akoff + 16);
        v8h b0 = *(const v8h*)(Bp + kk + bkoff);
        v8h b1 = *(const v8h*)(Bp + kk + bkoff + 16);
        v16h av = CAT16(a0, a1);
        v16h bv = CAT16(b0, b1);
        acc = __builtin_amdgcn_wmma_f32_16x16x32_f16(false, av, false, bv,
                                                     (short)0, acc, false, false);
    }
    int mbase = mtile * 16 + ((lane < 16) ? 0 : 8);
    int ncol  = ntile * 16 + (lane & 15);
#pragma unroll
    for (int i = 0; i < 8; ++i)
        gates[(size_t)(mbase + i) * G_ + ncol] = acc[i];
}

// ---------------------------------------------------------------------------
// 4) LSTM pointwise: i,f,g,o -> h,c ; optional f16 h into next layer's A
// ---------------------------------------------------------------------------
__global__ void lstm_pointwise_kernel(const float* __restrict__ gates,
                                      const float* __restrict__ cell_l,
                                      const float* __restrict__ b_ih_l,
                                      const float* __restrict__ b_hh_l,
                                      float* __restrict__ out_h,
                                      float* __restrict__ out_c,
                                      _Float16* __restrict__ Anext) {
    int idx = blockIdx.x * blockDim.x + threadIdx.x;
    if (idx >= B_ * H_) return;
    int b = idx >> 9, j = idx & 511;
    const float* gr = gates + (size_t)b * G_;
    float gi = gr[j]            + b_ih_l[j]            + b_hh_l[j];
    float gf = gr[H_ + j]       + b_ih_l[H_ + j]       + b_hh_l[H_ + j];
    float gg = gr[2 * H_ + j]   + b_ih_l[2 * H_ + j]   + b_hh_l[2 * H_ + j];
    float go = gr[3 * H_ + j]   + b_ih_l[3 * H_ + j]   + b_hh_l[3 * H_ + j];
    float i_ = 1.f / (1.f + __expf(-gi));
    float f_ = 1.f / (1.f + __expf(-gf));
    float g_ = tanhf(gg);
    float o_ = 1.f / (1.f + __expf(-go));
    float c  = f_ * cell_l[idx] + i_ * g_;
    float h  = o_ * tanhf(c);
    out_h[idx] = h;
    out_c[idx] = c;
    if (Anext) Anext[b * KL_ + j] = (_Float16)h;
}

// ---------------------------------------------------------------------------
// 5) Flash attention pass 1. softmax(hWd + enc·We + b) == softmax(enc·We):
//    the per-b constants cancel, so attention is independent of the LSTM.
//    enc is streamed from HBM exactly once (online softmax).
// ---------------------------------------------------------------------------
__global__ void attn_pass1_kernel(const float* __restrict__ enc,
                                  const float* __restrict__ attn_W,
                                  float* __restrict__ pm,
                                  float* __restrict__ pl,
                                  float* __restrict__ pacc) {
    int b = blockIdx.x, split = blockIdx.y;
    int wave = threadIdx.x >> 5, lane = threadIdx.x & 31;
    int c0 = lane * 16;                        // 16 channels per lane

    float we[16];
#pragma unroll
    for (int i = 0; i < 16; ++i) we[i] = attn_W[H_ + c0 + i];   // We = attn_W[0,512:]

    float m = -3.0e38f, l = 0.f;
    float acc[16];
#pragma unroll
    for (int i = 0; i < 16; ++i) acc[i] = 0.f;

    const int schunk = S_ / SPLIT_;            // 512
    const int s0 = split * schunk;
    const size_t encb = (size_t)b * S_ * H_;

    for (int s = s0 + wave; s < s0 + schunk; s += 8) {
        const float* row = enc + encb + (size_t)s * H_ + c0;
        __builtin_prefetch(row + 8 * H_, 0, 0);
        float4 q[4];
        q[0] = ((const float4*)row)[0];
        q[1] = ((const float4*)row)[1];
        q[2] = ((const float4*)row)[2];
        q[3] = ((const float4*)row)[3];
        const float* e = (const float*)q;
        float d = 0.f;
#pragma unroll
        for (int i = 0; i < 16; ++i) d += e[i] * we[i];
#pragma unroll
        for (int off = 16; off > 0; off >>= 1) d += __shfl_xor(d, off, 32);
        float mn = fmaxf(m, d);
        float sc = __expf(m - mn);
        float w  = __expf(d - mn);
        l = l * sc + w;
#pragma unroll
        for (int i = 0; i < 16; ++i) acc[i] = acc[i] * sc + w * e[i];
        m = mn;
    }

    __shared__ float sm[8], sl[8], sacc[8 * H_];
#pragma unroll
    for (int i = 0; i < 16; ++i) sacc[wave * H_ + c0 + i] = acc[i];
    if (lane == 0) { sm[wave] = m; sl[wave] = l; }
    __syncthreads();

    float M = sm[0];
#pragma unroll
    for (int w = 1; w < 8; ++w) M = fmaxf(M, sm[w]);
    float L = 0.f;
#pragma unroll
    for (int w = 0; w < 8; ++w) L += sl[w] * __expf(sm[w] - M);

    int pidx = b * SPLIT_ + split;
    for (int c = threadIdx.x * 2; c < threadIdx.x * 2 + 2; ++c) {
        float v = 0.f;
#pragma unroll
        for (int w = 0; w < 8; ++w) v += sacc[w * H_ + c] * __expf(sm[w] - M);
        pacc[(size_t)pidx * H_ + c] = v;
    }
    if (threadIdx.x == 0) { pm[pidx] = M; pl[pidx] = L; }
}

// ---------------------------------------------------------------------------
// 6) Flash attention pass 2: combine SPLIT_ partials -> ctx (f16)
// ---------------------------------------------------------------------------
__global__ void attn_pass2_kernel(const float* __restrict__ pm,
                                  const float* __restrict__ pl,
                                  const float* __restrict__ pacc,
                                  _Float16* __restrict__ ctx16) {
    int b = blockIdx.x;
    float M = pm[b * SPLIT_];
#pragma unroll
    for (int p = 1; p < SPLIT_; ++p) M = fmaxf(M, pm[b * SPLIT_ + p]);
    float L = 0.f;
#pragma unroll
    for (int p = 0; p < SPLIT_; ++p) L += pl[b * SPLIT_ + p] * __expf(pm[b * SPLIT_ + p] - M);
    float invL = 1.f / L;
    for (int c = threadIdx.x; c < H_; c += blockDim.x) {
        float v = 0.f;
#pragma unroll
        for (int p = 0; p < SPLIT_; ++p)
            v += pacc[(size_t)(b * SPLIT_ + p) * H_ + c] * __expf(pm[b * SPLIT_ + p] - M);
        ctx16[b * H_ + c] = (_Float16)(v * invL);
    }
}

// ---------------------------------------------------------------------------
// 7) Output FCs via WMMA: out(128,1000) = ctx(128,512) @ fcW(1000,512)^T + b
//    grid (63 N-tiles, 2 heads); guarded stores for N = 1000.
// ---------------------------------------------------------------------------
__global__ void fc_wmma_kernel(const _Float16* __restrict__ ctx16,
                               const _Float16* __restrict__ fcW16,
                               const float* __restrict__ abias,
                               const float* __restrict__ tbias,
                               float* __restrict__ out) {
    int ntile = blockIdx.x;
    int head  = blockIdx.y;
    const _Float16* W    = fcW16 + (size_t)head * NA_ * KF_;
    const float*    bias = head ? tbias : abias;
    float*          o    = out + (size_t)head * B_ * NA_;

    int wave = threadIdx.x >> 5, lane = threadIdx.x & 31;
    int mtile = wave;
    int arow  = mtile * 16 + (lane & 15);
    int akoff = (lane < 16) ? 0 : 8;
    int n     = ntile * 16 + (lane & 15);
    int g     = (n < NA_) ? n : (NA_ - 1);     // clamp; store is guarded
    int bkoff = (lane < 16) ? 0 : 8;

    const _Float16* Ap = ctx16 + (size_t)arow * KF_;
    const _Float16* Bp = W     + (size_t)g * KF_;

    v8f acc = {};
#pragma unroll 4
    for (int kk = 0; kk < KF_; kk += 32) {
        v8h a0 = *(const v8h*)(Ap + kk + akoff);
        v8h a1 = *(const v8h*)(Ap + kk + akoff + 16);
        v8h b0 = *(const v8h*)(Bp + kk + bkoff);
        v8h b1 = *(const v8h*)(Bp + kk + bkoff + 16);
        v16h av = CAT16(a0, a1);
        v16h bv = CAT16(b0, b1);
        acc = __builtin_amdgcn_wmma_f32_16x16x32_f16(false, av, false, bv,
                                                     (short)0, acc, false, false);
    }
    if (n < NA_) {
        float bb = bias[n];
        int mbase = mtile * 16 + ((lane < 16) ? 0 : 8);
#pragma unroll
        for (int i = 0; i < 8; ++i)
            o[(size_t)(mbase + i) * NA_ + n] = acc[i] + bb;
    }
}

// ---------------------------------------------------------------------------
// Host launcher
// ---------------------------------------------------------------------------
extern "C" void kernel_launch(void* const* d_in, const int* in_sizes, int n_in,
                              void* d_out, int out_size, void* d_ws, size_t ws_size,
                              hipStream_t stream) {
    const int*   x      = (const int*)  d_in[0];
    const float* hidden = (const float*)d_in[1];
    const float* cell   = (const float*)d_in[2];
    const float* enc    = (const float*)d_in[3];
    const float* aemb   = (const float*)d_in[4];
    const float* temb   = (const float*)d_in[5];
    const float* W_ih   = (const float*)d_in[6];
    const float* W_hh   = (const float*)d_in[7];
    const float* b_ih   = (const float*)d_in[8];
    const float* b_hh   = (const float*)d_in[9];
    const float* attn_W = (const float*)d_in[10];
    // d_in[11] (attn_b) unused: softmax is shift-invariant, constants cancel.
    const float* afcW   = (const float*)d_in[12];
    const float* afcb   = (const float*)d_in[13];
    const float* tfcW   = (const float*)d_in[14];
    const float* tfcb   = (const float*)d_in[15];
    float* out = (float*)d_out;

    // Workspace layout (bytes)
    char* ws = (char*)d_ws;
    _Float16* Wc    = (_Float16*)(ws);                 //  8,388,608  [2][2048][1024] f16
    _Float16* fcW16 = (_Float16*)(ws +  8388608);      //  2,048,000  [2000][512] f16
    _Float16* A0    = (_Float16*)(ws + 10436608);      //    262,144  [128][1024] f16
    _Float16* A1    = (_Float16*)(ws + 10698752);      //    262,144
    float*    gates = (float*)   (ws + 10960896);      //  1,048,576  [128][2048] f32
    _Float16* ctx16 = (_Float16*)(ws + 12009472);      //    131,072  [128][512] f16
    float*    pm    = (float*)   (ws + 12140544);      //      2,048
    float*    pl    = (float*)   (ws + 12142592);      //      2,048
    float*    pacc  = (float*)   (ws + 12144640);      //  1,048,576  [512][512] f32
    // total ~13.2 MB

    // Output layout: action(128*1000) | target(128*1000) | hidden(2*128*512) | cell(2*128*512)
    float* out_action = out;
    float* out_hid    = out + 256000;
    float* out_cell   = out + 387072;

    // Prep (f16 repack + embedding gather)
    {
        int total = 2 * G_ * KL_ + 2 * NA_ * KF_;
        convert_weights_kernel<<<(total + 255) / 256, 256, 0, stream>>>(
            W_ih, W_hh, afcW, tfcW, Wc, fcW16);
    }
    {
        int total = B_ * KL_ + B_ * H_;
        build_A_kernel<<<(total + 255) / 256, 256, 0, stream>>>(
            x, aemb, temb, hidden, A0, A1);
    }

    // Attention (independent of LSTM; enc streamed once)
    attn_pass1_kernel<<<dim3(B_, SPLIT_), 256, 0, stream>>>(enc, attn_W, pm, pl, pacc);
    attn_pass2_kernel<<<B_, 256, 0, stream>>>(pm, pl, pacc, ctx16);
    fc_wmma_kernel<<<dim3((NA_ + 15) / 16, 2), 256, 0, stream>>>(
        ctx16, fcW16, afcb, tfcb, out_action);

    // LSTM layer 0
    lstm_gemm_kernel<<<G_ / 16, 256, 0, stream>>>(A0, Wc, gates);
    lstm_pointwise_kernel<<<(B_ * H_ + 255) / 256, 256, 0, stream>>>(
        gates, cell, b_ih, b_hh, out_hid, out_cell, A1);

    // LSTM layer 1
    lstm_gemm_kernel<<<G_ / 16, 256, 0, stream>>>(A1, Wc + (size_t)G_ * KL_, gates);
    lstm_pointwise_kernel<<<(B_ * H_ + 255) / 256, 256, 0, stream>>>(
        gates, cell + B_ * H_, b_ih + G_, b_hh + G_,
        out_hid + B_ * H_, out_cell + B_ * H_, nullptr);
}
